// LinearSystem_34084860461065
// MI455X (gfx1250) — matrix operand order repaired and checked by reading, not compiled, not run on hardware
//
#include <hip/hip_runtime.h>

// ---------------------------------------------------------------------------
// Linear state-space simulation on MI455X (gfx1250), wave32 + WMMA bf16.
//   states[b,t] = states[b,t-1] @ F^T + inputs[b,t] @ Bm^T + Wn[b,t] @ SW^T
//   obs[b,t]    = states[b,t] @ H^T + Vn[b,t] @ SV^T
// All matmuls use v_wmma_f32_16x16x32_bf16 (bf16 inputs, f32 accumulation).
// ---------------------------------------------------------------------------

typedef __attribute__((ext_vector_type(16))) __bf16 v16bf;
typedef __attribute__((ext_vector_type(8)))  __bf16 v8bf;
typedef __attribute__((ext_vector_type(8)))  float  v8f;
typedef __attribute__((ext_vector_type(4)))  float  v4f;

// Explicit global (addrspace 1) pointers, used to re-assert "global" after
// laundering a pointer through inline asm (otherwise the backend emits
// flat_load, which ticks both LOADcnt and DScnt).
typedef const __attribute__((address_space(1))) __bf16* gptr_bf;
typedef const __attribute__((address_space(1))) v8bf*   gptr_v8bf;

#define SDIM 512
#define IDIM 256
#define ODIM 256
#define NB   64
#define NT   1024
#define ROWS (NB * NT)
#define SPAD 520   // LDS row stride (bf16 elems): 512 + 8 pad to spread banks

// ---- fp32 -> bf16 weight conversion -------------------------------------
__global__ void cvt_bf16_kernel(const float* __restrict__ src,
                                __bf16* __restrict__ dst, int n) {
    int i = blockIdx.x * blockDim.x + threadIdx.x;
    if (i < n) dst[i] = (__bf16)src[i];
}

__device__ inline v16bf cat8(v8bf a, v8bf b) {
    return __builtin_shufflevector(a, b, 0, 1, 2, 3, 4, 5, 6, 7,
                                         8, 9, 10, 11, 12, 13, 14, 15);
}

// A-fragment (16x32 bf16, M x K) from an fp32 row.  Lane holds row m=lane&15;
// elements 0..7  <- A[m][k0 + 16*hf + 0..7]
// elements 8..15 <- A[m][k0 + 16*hf + 8 + 0..7]   (contiguous k, matches B)
__device__ inline v16bf load_a_f32(const float* __restrict__ row, int hf) {
    const v4f* p = (const v4f*)(row + 16 * hf);
    const v4f* q = (const v4f*)(row + 16 * hf + 8);
    v4f x0 = p[0], x1 = p[1], x2 = q[0], x3 = q[1];
    v16bf a;
#pragma unroll
    for (int i = 0; i < 4; ++i) {
        a[i]      = (__bf16)x0[i];
        a[4 + i]  = (__bf16)x1[i];
        a[8 + i]  = (__bf16)x2[i];
        a[12 + i] = (__bf16)x3[i];
    }
    return a;
}

// B-fragment (32x16 bf16, K x N) for C = X @ W^T with W row-major [n][k]:
// lane holds weight-row n = n0+(lane&15), elements e <- W[n][k0 + 16*hf + e].
__device__ inline v16bf load_b_bf16(const __bf16* p) {
    const v8bf* q = (const v8bf*)p;
    return cat8(q[0], q[1]);
}

__device__ inline v16bf load_b_bf16_g(gptr_bf p) {
    gptr_v8bf q = (gptr_v8bf)p;
    return cat8(q[0], q[1]);
}

__device__ inline v8f wmma_bf16(v16bf a, v16bf b, v8f c) {
    return __builtin_amdgcn_wmma_f32_16x16x32_bf16(
        false, a, false, b, (short)0, c, false, false);
}

// Accumulate a 32x64 macro-tile (2 M-subtiles x 4 N-subtiles) of
// A(f32)[rows x K] @ W(bf16)[N x K]^T with a 2-stage K pipeline.
template <int K>
__device__ inline void mm_macc(const float* __restrict__ arow,   // A + (r0+ln)*K
                               const __bf16* __restrict__ wrow,  // W + (n0+ln)*K + 16*hf
                               int hf, v8f acc[2][4]) {
    constexpr int NC = K / 32;
    v16bf a[2][2], b[2][4];

    auto ld = [&](int st, int kc) {
        a[st][0] = load_a_f32(arow + kc, hf);
        a[st][1] = load_a_f32(arow + (size_t)16 * K + kc, hf);
#pragma unroll
        for (int j = 0; j < 4; ++j)
            b[st][j] = load_b_bf16(wrow + (size_t)j * 16 * K + kc);
    };

    ld(0, 0);
#pragma unroll
    for (int i = 0; i < NC; ++i) {
        const int cur = i & 1;
        if (i + 1 < NC) ld(cur ^ 1, (i + 1) * 32);  // prefetch next chunk
#pragma unroll
        for (int mi = 0; mi < 2; ++mi)
#pragma unroll
            for (int j = 0; j < 4; ++j)
                acc[mi][j] = wmma_bf16(a[cur][mi], b[cur][j], acc[mi][j]);
    }
}

// ---- Phases 1 & 3: C[r,n] = A1@W1^T + A2@W2^T ---------------------------
// One 32x64 macro-tile per wave, 8 waves per block, exact grid coverage.
template <int K1, int K2>
__global__ void __launch_bounds__(256) gemm2_kernel(
    const float* __restrict__ A1, const __bf16* __restrict__ W1,
    const float* __restrict__ A2, const __bf16* __restrict__ W2,
    float* __restrict__ C, int N, int tiles_n) {
    const int wave = threadIdx.x >> 5;
    const int lane = threadIdx.x & 31;
    const int tile = blockIdx.x * 8 + wave;
    const int r0 = (tile / tiles_n) * 32;
    const int n0 = (tile % tiles_n) * 64;
    const int ln = lane & 15, hf = lane >> 4;

    v8f acc[2][4];
#pragma unroll
    for (int mi = 0; mi < 2; ++mi)
#pragma unroll
        for (int j = 0; j < 4; ++j) acc[mi][j] = (v8f){};

    mm_macc<K1>(A1 + (size_t)(r0 + ln) * K1,
                W1 + (size_t)(n0 + ln) * K1 + 16 * hf, hf, acc);
    mm_macc<K2>(A2 + (size_t)(r0 + ln) * K2,
                W2 + (size_t)(n0 + ln) * K2 + 16 * hf, hf, acc);

    // f32 C/D layout: lane holds column (n-subtile)+ln, rows 8*hf + r
#pragma unroll
    for (int mi = 0; mi < 2; ++mi) {
        float* crow = C + (size_t)(r0 + mi * 16 + 8 * hf) * N + n0 + ln;
#pragma unroll
        for (int j = 0; j < 4; ++j)
#pragma unroll
            for (int r = 0; r < 8; ++r)
                crow[(size_t)r * N + j * 16] = acc[mi][j][r];
    }
}

// ---- Phase 2: sequential scan over T ------------------------------------
// grid = 4 blocks x 1024 threads; block owns 16 batch rows (independent
// recurrences).  State tile lives in LDS as bf16; each of the 32 waves owns
// 16 state columns; drive is read from (and states written back to) the
// output buffer in place.  F streams from L2 (512 KB bf16, fully resident);
// an empty asm on the F pointer stops the compiler from hoisting F loads out
// of the t-loop (which previously caused catastrophic scratch spills), and
// the addrspace(1) cast keeps those loads on the global path (LOADcnt only,
// decoupled from the LDS/DScnt A-stream).
// B-fragments prefetch 4 chunks ahead; A-fragments (LDS) 2 ahead.
__global__ void __launch_bounds__(1024) scan_kernel(
    const float* __restrict__ state0, const __bf16* __restrict__ Fbf,
    float* __restrict__ states) {
    __shared__ __bf16 s_bf[16 * SPAD];
    const int b0  = blockIdx.x * 16;
    const int tid = threadIdx.x;

    // seed s_{-1} = initial state (f32 -> bf16)
    for (int i = tid; i < 16 * SDIM; i += 1024) {
        int m = i >> 9, k = i & (SDIM - 1);
        s_bf[m * SPAD + k] = (__bf16)state0[(size_t)(b0 + m) * SDIM + k];
    }
    __syncthreads();

    const int wave = tid >> 5, lane = tid & 31;
    const int ln = lane & 15, hf = lane >> 4;
    const int col = wave * 16 + ln;
    const __bf16* frow = Fbf + (size_t)col * SDIM + 16 * hf;  // F[n][*]
    const __bf16* arow = &s_bf[ln * SPAD];

    // 32-bit offsets into `states` for (row, t=0, col); += SDIM each step
    unsigned goff[8];
#pragma unroll
    for (int r = 0; r < 8; ++r)
        goff[r] = (unsigned)((b0 + 8 * hf + r) * (NT * SDIM) + col);
    const int lcol = (8 * hf) * SPAD + col;  // LDS write base for this lane

    for (int t = 0; t < NT; ++t) {
        // Defeat loop-invariant hoisting of the F loads: make the F pointer
        // opaque per iteration, then re-assert global address space.
        unsigned long long fp_ = (unsigned long long)frow;
        asm volatile("" : "+v"(fp_));
        gptr_bf fptr = (gptr_bf)fp_;

        // start accumulator from drive_t (written by phase 1 into `states`)
        v8f acc;
#pragma unroll
        for (int r = 0; r < 8; ++r) acc[r] = states[goff[r]];

        // acc += s_{t-1} @ F^T : K = 512, 16 bf16 WMMAs.
        // B prefetch depth 4 (L2 latency), A prefetch depth 2 (LDS).
        v16bf a[2], b[4];
        auto lda = [&](int st, int kc) {
            a[st] = cat8(*(const v8bf*)(arow + kc + 16 * hf),
                         *(const v8bf*)(arow + kc + 16 * hf + 8));
        };
        auto ldb = [&](int st, int kc) { b[st] = load_b_bf16_g(fptr + kc); };

        ldb(0, 0); ldb(1, 32); ldb(2, 64); ldb(3, 96);
        lda(0, 0); lda(1, 32);
#pragma unroll
        for (int i = 0; i < SDIM / 32; ++i) {
            if (i + 4 < SDIM / 32) ldb((i + 4) & 3, (i + 4) * 32);
            if (i + 2 < SDIM / 32) lda((i + 2) & 1, (i + 2) * 32);
            acc = wmma_bf16(a[i & 1], b[i & 3], acc);
        }

        __syncthreads();  // all waves done reading s_{t-1}
#pragma unroll
        for (int r = 0; r < 8; ++r) {
            states[goff[r]] = acc[r];
            s_bf[lcol + r * SPAD] = (__bf16)acc[r];
            goff[r] += SDIM;
        }
        __syncthreads();  // s_t visible to all waves
    }
}

// ---------------------------------------------------------------------------
extern "C" void kernel_launch(void* const* d_in, const int* in_sizes, int n_in,
                              void* d_out, int out_size, void* d_ws,
                              size_t ws_size, hipStream_t stream) {
    (void)in_sizes; (void)n_in; (void)out_size; (void)ws_size;
    const float* state  = (const float*)d_in[0];
    const float* inputs = (const float*)d_in[1];
    const float* Wn     = (const float*)d_in[2];
    const float* Vn     = (const float*)d_in[3];
    const float* F      = (const float*)d_in[4];
    const float* Bm     = (const float*)d_in[5];
    const float* H      = (const float*)d_in[6];
    const float* SW     = (const float*)d_in[7];
    const float* SV     = (const float*)d_in[8];

    float* states = (float*)d_out;                    // (64,1024,512)
    float* obs    = states + (size_t)ROWS * SDIM;     // (64,1024,256)

    // bf16 weights in workspace (~1.7 MB total)
    __bf16* Fbf  = (__bf16*)d_ws;
    __bf16* Bmbf = Fbf  + SDIM * SDIM;
    __bf16* SWbf = Bmbf + SDIM * IDIM;
    __bf16* Hbf  = SWbf + SDIM * SDIM;
    __bf16* SVbf = Hbf  + ODIM * SDIM;

    cvt_bf16_kernel<<<(SDIM * SDIM) / 256, 256, 0, stream>>>(F,  Fbf,  SDIM * SDIM);
    cvt_bf16_kernel<<<(SDIM * IDIM) / 256, 256, 0, stream>>>(Bm, Bmbf, SDIM * IDIM);
    cvt_bf16_kernel<<<(SDIM * SDIM) / 256, 256, 0, stream>>>(SW, SWbf, SDIM * SDIM);
    cvt_bf16_kernel<<<(ODIM * SDIM) / 256, 256, 0, stream>>>(H,  Hbf,  ODIM * SDIM);
    cvt_bf16_kernel<<<(ODIM * ODIM) / 256, 256, 0, stream>>>(SV, SVbf, ODIM * ODIM);

    // Phase 1: drive = inputs @ Bm^T + Wn @ SW^T  -> states buffer
    // macro-tiles = (65536/32) * (512/64) = 16384; 8 per block
    gemm2_kernel<IDIM, SDIM><<<16384 / 8, 256, 0, stream>>>(
        inputs, Bmbf, Wn, SWbf, states, SDIM, SDIM / 64);

    // Phase 2: in-place sequential scan (4 independent 16-row chains)
    scan_kernel<<<NB / 16, 1024, 0, stream>>>(state, Fbf, states);

    // Phase 3: obs = states @ H^T + Vn @ SV^T
    // macro-tiles = (65536/32) * (256/64) = 8192; 8 per block
    gemm2_kernel<SDIM, IDIM><<<8192 / 8, 256, 0, stream>>>(
        states, Hbf, Vn, SVbf, obs, ODIM, ODIM / 64);
}
